// VisionMamba_65180423685238
// MI455X (gfx1250) — matrix operand order repaired and checked by reading, not compiled, not run on hardware
//
#include <hip/hip_runtime.h>
#include <hip/hip_bf16.h>
#include <math.h>

// ---- model dims (fixed by the reference) ----
#define BATCH   8
#define LSEQ    196            // (224/16)^2
#define BL      (BATCH*LSEQ)   // 1568
#define DM      192            // d_model
#define DI      384            // d_inner
#define DSTATE  16
#define DTR     12
#define DXP     44             // dt_rank + 2*d_state
#define DEPTHN  24
#define IMGSZ   224
#define KPE     768            // 3*16*16 patch-embed K

typedef __attribute__((ext_vector_type(16))) __bf16 v16bf;
typedef __attribute__((ext_vector_type(8)))  float  v8f;

__device__ __forceinline__ __bf16 f2bf(float f) { return (__bf16)f; }

// ---------------------------------------------------------------------------
// Fast WMMA GEMM: C[M,N] = A[M,K] * W[N,K]^T (+bias[N])
// Requirements: M%16==0, N%64==0, K%32==0 (patch/in_proj/outproj GEMMs).
// One wave -> 16(M) x 64(N) tile, 4 accumulators, A fragment reused.
// LDS tiles are stored unpermuted [row][k] so every fragment is one contiguous
// 32-byte LDS read (2x ds_load_b128 per fragment).
//   A[16][32]: stored [m][k], k = 16*half + j -> lane reads
//              &As[mr][half*16].
//   B[64][32]: stored [n][k] -> fragment w reads &Bs[w*16+mr][half*16].
// ---------------------------------------------------------------------------
__global__ void k_gemm_fast(const float* __restrict__ A, long lda,
                            const float* __restrict__ W, long ldw,
                            float* __restrict__ C, long ldc,
                            const float* __restrict__ bias,
                            int K) {
    __shared__ __align__(32) __bf16 As[16][32];
    __shared__ __align__(32) __bf16 Bs[64][32];

    const int tid  = threadIdx.x;            // 0..31, one wave
    const int m0   = blockIdx.x * 16;
    const int n0   = blockIdx.y * 64;
    const int half = (tid >> 4) & 1;
    const int mr   = tid & 15;

    v8f acc0 = {}, acc1 = {}, acc2 = {}, acc3 = {};

    for (int kk = 0; kk < K; kk += 32) {
        if (kk + 32 < K)   // prefetch next weight K-block (global_prefetch_b8)
            __builtin_prefetch(&W[(long)n0 * ldw + kk + 32], 0, 0);

        // A tile: 512 floats = 128 float4; 4 per thread (coalesced b128).
        // Written unpermuted: As[r][k], k in natural order.
#pragma unroll
        for (int i = 0; i < 4; ++i) {
            int e4 = tid + 32 * i;           // 0..127
            int r  = e4 >> 3;                // 8 float4 per 32-elem row
            int c4 = (e4 & 7) * 4;
            const float4 v = *reinterpret_cast<const float4*>(
                &A[(long)(m0 + r) * lda + kk + c4]);
            int g  = c4 >> 3;
            int gp = ((g & 1) << 1) | (g >> 1);
            int kp = c4;
            As[r][kp + 0] = f2bf(v.x);
            As[r][kp + 1] = f2bf(v.y);
            As[r][kp + 2] = f2bf(v.z);
            As[r][kp + 3] = f2bf(v.w);
        }
        // B tile: 64 n-rows x 32 k = 512 float4; 16 per thread, stored [n][k]
#pragma unroll
        for (int i = 0; i < 16; ++i) {
            int e4 = tid + 32 * i;           // 0..511
            int nr = e4 >> 3;                // 0..63
            int c4 = (e4 & 7) * 4;
            const float4 v = *reinterpret_cast<const float4*>(
                &W[(long)(n0 + nr) * ldw + kk + c4]);
            Bs[nr][c4 + 0] = f2bf(v.x);
            Bs[nr][c4 + 1] = f2bf(v.y);
            Bs[nr][c4 + 2] = f2bf(v.z);
            Bs[nr][c4 + 3] = f2bf(v.w);
        }
        __syncthreads();

        const v16bf af = *reinterpret_cast<const v16bf*>(&As[mr][half * 16]);
        v16bf bf;
        bf = *reinterpret_cast<const v16bf*>(&Bs[mr][half * 16]);
        acc0 = __builtin_amdgcn_wmma_f32_16x16x32_bf16(false, af, false, bf,
                                                       (short)0, acc0, false, false);
        bf = *reinterpret_cast<const v16bf*>(&Bs[16 + mr][half * 16]);
        acc1 = __builtin_amdgcn_wmma_f32_16x16x32_bf16(false, af, false, bf,
                                                       (short)0, acc1, false, false);
        bf = *reinterpret_cast<const v16bf*>(&Bs[32 + mr][half * 16]);
        acc2 = __builtin_amdgcn_wmma_f32_16x16x32_bf16(false, af, false, bf,
                                                       (short)0, acc2, false, false);
        bf = *reinterpret_cast<const v16bf*>(&Bs[48 + mr][half * 16]);
        acc3 = __builtin_amdgcn_wmma_f32_16x16x32_bf16(false, af, false, bf,
                                                       (short)0, acc3, false, false);
        __syncthreads();
    }

    // D layout: VGPR r -> M = r + 8*half, N = lane&15 (per 16-col group)
    const int mbase = m0 + half * 8;
    {
        int n = n0 + mr;          float bv = bias ? bias[n] : 0.0f;
#pragma unroll
        for (int r = 0; r < 8; ++r) C[(long)(mbase + r) * ldc + n] = acc0[r] + bv;
    }
    {
        int n = n0 + 16 + mr;     float bv = bias ? bias[n] : 0.0f;
#pragma unroll
        for (int r = 0; r < 8; ++r) C[(long)(mbase + r) * ldc + n] = acc1[r] + bv;
    }
    {
        int n = n0 + 32 + mr;     float bv = bias ? bias[n] : 0.0f;
#pragma unroll
        for (int r = 0; r < 8; ++r) C[(long)(mbase + r) * ldc + n] = acc2[r] + bv;
    }
    {
        int n = n0 + 48 + mr;     float bv = bias ? bias[n] : 0.0f;
#pragma unroll
        for (int r = 0; r < 8; ++r) C[(long)(mbase + r) * ldc + n] = acc3[r] + bv;
    }
}

// ---------------------------------------------------------------------------
// Edge WMMA GEMM (guarded, 16x16 tile): used for the ragged GEMMs
// (xproj N=44, dtproj K=12). Same unpermuted [row][k] LDS fragment layout.
// ---------------------------------------------------------------------------
__global__ void k_gemm_edge(const float* __restrict__ A, long lda,
                            const float* __restrict__ W, long ldw,
                            float* __restrict__ C, long ldc,
                            int M, int N, int K) {
    __shared__ __align__(32) __bf16 As[16][32];   // [m][k]
    __shared__ __align__(32) __bf16 Bs[16][32];   // [n][k]

    const int tid  = threadIdx.x;
    const int m0   = blockIdx.x * 16;
    const int n0   = blockIdx.y * 16;
    const int half = (tid >> 4) & 1;
    const int mr   = tid & 15;

    v8f acc = {};

    for (int kk = 0; kk < K; kk += 32) {
        for (int e = tid; e < 512; e += 32) {
            int r = e >> 5, k = e & 31;
            int g = k >> 3, gp = ((g & 1) << 1) | (g >> 1);
            int kp = k;
            int gm = m0 + r, gk = kk + k;
            As[r][kp] = (gm < M && gk < K) ? f2bf(A[(long)gm * lda + gk])
                                           : (__bf16)0.0f;
            int gn = n0 + r;
            Bs[r][k] = (gn < N && gk < K) ? f2bf(W[(long)gn * ldw + gk])
                                          : (__bf16)0.0f;
        }
        __syncthreads();
        const v16bf af = *reinterpret_cast<const v16bf*>(&As[mr][half * 16]);
        // B fragment element j <-> (k = j + half*16, n = mr): with [n][k]
        // storage this is contiguous, in the same un-permuted k order as A; since the
        // guarded path zero-pads, use the same contiguous read then rely on
        // matching store-side order above.
        v16bf bf;
#pragma unroll
        for (int j = 0; j < 16; ++j) bf[j] = Bs[mr][j + half * 16];
        acc = __builtin_amdgcn_wmma_f32_16x16x32_bf16(false, af, false, bf,
                                                      (short)0, acc, false, false);
        __syncthreads();
    }

    int n = n0 + mr;
    if (n < N) {
#pragma unroll
        for (int r = 0; r < 8; ++r) {
            int m = m0 + r + half * 8;
            if (m < M) C[(long)m * ldc + n] = acc[r];
        }
    }
}

// ---------------------------------------------------------------------------
// im2col for the 16x16 stride-16 patch embed: Acol[BL][768]
// ---------------------------------------------------------------------------
__global__ void k_im2col(const float* __restrict__ x, float* __restrict__ Acol) {
    int idx = blockIdx.x * blockDim.x + threadIdx.x;
    if (idx >= BL * KPE) return;
    int m = idx / KPE, k = idx % KPE;
    int b = m / LSEQ, l = m % LSEQ;
    int py = l / 14, px = l % 14;
    int c = k / 256, rem = k % 256;
    int ky = rem / 16, kx = rem % 16;
    int iy = py * 16 + ky, ix = px * 16 + kx;
    Acol[idx] = x[((long)b * 3 + c) * IMGSZ * IMGSZ + (long)iy * IMGSZ + ix];
}

// residual = h  (first) or residual += h
__global__ void k_addres(float* __restrict__ res, const float* __restrict__ h,
                         int first) {
    int idx = blockIdx.x * blockDim.x + threadIdx.x;
    if (idx >= BL * DM) return;
    res[idx] = first ? h[idx] : res[idx] + h[idx];
}

// LayerNorm over D_MODEL=192; one block per row, blockDim = 192
__global__ void k_layernorm(const float* __restrict__ in,
                            const float* __restrict__ w,
                            const float* __restrict__ b,
                            float* __restrict__ out) {
    __shared__ float buf[DM];
    int row = blockIdx.x, tid = threadIdx.x;
    float v = in[(long)row * DM + tid];
    buf[tid] = v;
    __syncthreads();
    for (int s = 128; s > 0; s >>= 1) {
        if (tid < s && tid + s < DM) buf[tid] += buf[tid + s];
        __syncthreads();
    }
    float mu = buf[0] / (float)DM;
    __syncthreads();
    float d = v - mu;
    buf[tid] = d * d;
    __syncthreads();
    for (int s = 128; s > 0; s >>= 1) {
        if (tid < s && tid + s < DM) buf[tid] += buf[tid + s];
        __syncthreads();
    }
    float rinv = rsqrtf(buf[0] / (float)DM + 1e-5f);
    out[(long)row * DM + tid] = d * rinv * w[tid] + b[tid];
}

// causal depthwise conv (K=4) over the xin half of xz, then SiLU
__global__ void k_convsilu(const float* __restrict__ xz,
                           const float* __restrict__ cw,
                           const float* __restrict__ cb,
                           float* __restrict__ xin) {
    int idx = blockIdx.x * blockDim.x + threadIdx.x;
    if (idx >= BL * DI) return;
    int row = idx / DI, d = idx % DI;
    int b = row / LSEQ, t = row % LSEQ;
    float s = cb[d];
#pragma unroll
    for (int j = 0; j < 4; ++j) {
        int tt = t - 3 + j;
        if (tt >= 0)
            s += cw[d * 4 + j] * xz[((long)(b * LSEQ + tt)) * (2 * DI) + d];
    }
    xin[idx] = s / (1.0f + __expf(-s));
}

// dt = softplus(dt_raw + dt_bias)   (in-place)
__global__ void k_softplus(float* __restrict__ dt, const float* __restrict__ dtb) {
    int idx = blockIdx.x * blockDim.x + threadIdx.x;
    if (idx >= BL * DI) return;
    float v = dt[idx] + dtb[idx % DI];
    dt[idx] = (v > 20.0f) ? v : log1pf(__expf(v));
}

// selective scan: grid = BATCH, block = DI; 16-state recurrence in registers.
// Per-timestep B/C vectors (32 contiguous floats) are broadcast to the block
// through LDS using the CDNA5 async global->LDS path (ASYNCcnt tracked).
__global__ void k_scan(const float* __restrict__ xin, const float* __restrict__ dt,
                       const float* __restrict__ xdbl,
                       const float* __restrict__ Alog, const float* __restrict__ Dp,
                       float* __restrict__ y) {
    int b = blockIdx.x, d = threadIdx.x;
    float Ad[DSTATE], st[DSTATE];
#pragma unroll
    for (int s = 0; s < DSTATE; ++s) {
        Ad[s] = -__expf(Alog[d * DSTATE + s]);
        st[s] = 0.0f;
    }
    float Dv = Dp[d];
    __shared__ float BC[2 * DSTATE];   // [0..15]=B, [16..31]=C
    for (int t = 0; t < LSEQ; ++t) {
        long row = (long)b * LSEQ + t;
        if (d < 2 * DSTATE) {
            // async copy: global (xdbl row, cols 12..43) -> LDS BC[0..31].
            // LDS operand: wave-relative offset = low 32 bits of generic ptr.
            unsigned lds = (unsigned)(unsigned long long)(&BC[d]);
            unsigned long long src =
                (unsigned long long)(const void*)(&xdbl[row * DXP + DTR + d]);
            asm volatile("global_load_async_to_lds_b32 %0, %1, off"
                         :: "v"(lds), "v"(src) : "memory");
        }
        asm volatile("s_wait_asynccnt 0x0" ::: "memory");
        __syncthreads();
        float u = xin[row * DI + d], dv = dt[row * DI + d];
        float du = dv * u, acc = 0.0f;
#pragma unroll
        for (int s = 0; s < DSTATE; ++s) {
            st[s] = __expf(dv * Ad[s]) * st[s] + du * BC[s];
            acc += st[s] * BC[DSTATE + s];
        }
        y[row * DI + d] = acc + u * Dv;
        __syncthreads();
    }
}

// yg = y * silu(z),  z = xz[..., 384:768]
__global__ void k_gate(const float* __restrict__ y, const float* __restrict__ xz,
                       float* __restrict__ yg) {
    int idx = blockIdx.x * blockDim.x + threadIdx.x;
    if (idx >= BL * DI) return;
    int row = idx / DI, d = idx % DI;
    float z = xz[(long)row * (2 * DI) + DI + d];
    yg[idx] = y[idx] * z / (1.0f + __expf(-z));
}

// ---------------------------------------------------------------------------
static inline void gemm(const float* A, long lda, const float* W, long ldw,
                        float* C, long ldc, const float* bias,
                        int M, int N, int K, hipStream_t s) {
    if ((K % 32 == 0) && (N % 64 == 0) && (M % 16 == 0)) {
        dim3 g(M / 16, N / 64);
        k_gemm_fast<<<g, 32, 0, s>>>(A, lda, W, ldw, C, ldc, bias, K);
    } else {
        dim3 g((M + 15) / 16, (N + 15) / 16);
        k_gemm_edge<<<g, 32, 0, s>>>(A, lda, W, ldw, C, ldc, M, N, K);
    }
}

extern "C" void kernel_launch(void* const* d_in, const int* in_sizes, int n_in,
                              void* d_out, int out_size, void* d_ws, size_t ws_size,
                              hipStream_t stream) {
    const float* x     = (const float*)d_in[0];
    const float* pe_w  = (const float*)d_in[1];   // (192, 768)
    const float* pe_b  = (const float*)d_in[2];
    const float* nw    = (const float*)d_in[3];   // (24,192)
    const float* nb    = (const float*)d_in[4];
    const float* ipw   = (const float*)d_in[5];   // (24,768,192)
    const float* cw    = (const float*)d_in[6];   // (24,384,4)
    const float* cb    = (const float*)d_in[7];   // (24,384)
    const float* xpw   = (const float*)d_in[8];   // (24,44,384)
    const float* dtw   = (const float*)d_in[9];   // (24,384,12)
    const float* dtb   = (const float*)d_in[10];  // (24,384)
    const float* Alog  = (const float*)d_in[11];  // (24,384,16)
    const float* Dp    = (const float*)d_in[12];  // (24,384)
    const float* opw   = (const float*)d_in[13];  // (24,192,384)
    const float* nfw   = (const float*)d_in[14];
    const float* nfb   = (const float*)d_in[15];
    float* out = (float*)d_out;

    // workspace partition (fp32), ~23.2 MB total
    float* ws   = (float*)d_ws;
    float* Acol = ws;  ws += (long)BL * KPE;      // im2col
    float* h    = ws;  ws += (long)BL * DM;
    float* res  = ws;  ws += (long)BL * DM;
    float* hn   = ws;  ws += (long)BL * DM;
    float* xz   = ws;  ws += (long)BL * 2 * DI;
    float* xin  = ws;  ws += (long)BL * DI;
    float* xdbl = ws;  ws += (long)BL * DXP;
    float* dt   = ws;  ws += (long)BL * DI;
    float* ybuf = ws;  ws += (long)BL * DI;
    float* yg   = ws;  ws += (long)BL * DI;

    const int EW_DI = (BL * DI + 255) / 256;   // 602112 elems
    const int EW_DM = (BL * DM + 255) / 256;   // 301056 elems

    // patch embed: im2col + GEMM(+bias)
    k_im2col<<<(BL * KPE + 255) / 256, 256, 0, stream>>>(x, Acol);
    gemm(Acol, KPE, pe_w, KPE, h, DM, pe_b, BL, DM, KPE, stream);

    for (int i = 0; i < DEPTHN; ++i) {
        k_addres<<<EW_DM, 256, 0, stream>>>(res, h, i == 0);
        k_layernorm<<<BL, DM, 0, stream>>>(res, nw + (long)i * DM, nb + (long)i * DM, hn);

        // xz = hn @ in_proj^T        (1568 x 768 x 192)  [fast]
        gemm(hn, DM, ipw + (long)i * 2 * DI * DM, DM, xz, 2 * DI, nullptr,
             BL, 2 * DI, DM, stream);

        // xin = silu(causal_dwconv(xz[:, :384]))
        k_convsilu<<<EW_DI, 256, 0, stream>>>(xz, cw + (long)i * DI * 4,
                                              cb + (long)i * DI, xin);

        // xdbl = xin @ xproj^T       (1568 x 44 x 384)   [edge]
        gemm(xin, DI, xpw + (long)i * DXP * DI, DI, xdbl, DXP, nullptr,
             BL, DXP, DI, stream);

        // dt_raw = xdbl[:, :12] @ dtproj^T  (1568 x 384 x 12) [edge]
        gemm(xdbl, DXP, dtw + (long)i * DI * DTR, DTR, dt, DI, nullptr,
             BL, DI, DTR, stream);
        k_softplus<<<EW_DI, 256, 0, stream>>>(dt, dtb + (long)i * DI);

        // selective scan
        k_scan<<<BATCH, DI, 0, stream>>>(xin, dt, xdbl,
                                         Alog + (long)i * DI * DSTATE,
                                         Dp + (long)i * DI, ybuf);

        // gate + out projection      (1568 x 192 x 384)  [fast]
        k_gate<<<EW_DI, 256, 0, stream>>>(ybuf, xz, yg);
        gemm(yg, DI, opw + (long)i * DM * DI, DI, h, DM, nullptr,
             BL, DM, DI, stream);
    }

    k_addres<<<EW_DM, 256, 0, stream>>>(res, h, 0);
    k_layernorm<<<BL, DM, 0, stream>>>(res, nfw, nfb, out);
}
